// Decoder_26577257628359
// MI455X (gfx1250) — matrix operand order repaired and checked by reading, not compiled, not run on hardware
//
#include <hip/hip_runtime.h>
#include <hip/hip_bf16.h>
#include <math.h>

typedef __bf16 bf16_t;
typedef __attribute__((ext_vector_type(16))) __bf16 v16bf;
typedef __attribute__((ext_vector_type(8)))  float  v8f;

#define B_    32
#define TENC  256
#define TDEC  800
#define ESPK  896
#define POSD  128
#define MAXLEN 100
#define NMEL  80
#define PRE_  256
#define IN1_  1280
#define KT_IN1 40      // 1280/32
#define KT_H   32      // 1024/32
#define NWG    16      // persistent workgroups in sequential kernel
#define PI_F   3.14159265358979323846f

// ---------------- fragment/swizzle helpers ----------------
// A-matrix 16x32 bf16 tile (ISA 7.12.2): lane = r + 16*k[3]; elem = (k&7) + 8*k[4]
__device__ __forceinline__ int a_off(int r, int k) {
  return (r + (((k >> 3) & 1) << 4)) * 16 + ((k & 7) + (((k >> 4) & 1) << 3));
}
// index into A-swizzled buffer laid out [mt][kt][lane][elem]
__device__ __forceinline__ size_t a_idx(int mt, int kt, int KT, int r, int k) {
  return ((size_t)(mt * KT + kt)) * 512 + a_off(r, k);
}

__device__ __forceinline__ v16bf ld_frag(const bf16_t* tile, int lane) {
  return *((const v16bf*)tile + lane);   // 32B per lane -> 2x global_load_b128
}

__device__ __forceinline__ v8f wmma_bf16(v16bf a, v16bf b, v8f c) {
  return __builtin_amdgcn_wmma_f32_16x16x32_bf16(false, a, false, b, (short)0, c,
                                                 false, false);
}

__device__ __forceinline__ float sigm(float x) { return 1.f / (1.f + expf(-x)); }

// ---------------- weight conversion + swizzle ----------------
// W is (N x K) row-major (dst B-fragment layout [nt][kt][lane][elem])
__global__ void k_bswz(const float* __restrict__ W, bf16_t* __restrict__ dst,
                       int N, int K, int stride, int colOff, int KT) {
  int idx = blockIdx.x * blockDim.x + threadIdx.x;
  int total = (N / 16) * KT * 512;
  if (idx >= total) return;
  int e = idx & 15, lane = (idx >> 4) & 31, tile = idx >> 9;
  int kt = tile % KT, nt = tile / KT;
  int n = nt * 16 + (lane & 15);
  int k = kt * 32 + ((lane >> 4) << 3) + (e & 7) + ((e >> 3) << 4);  // B layout: lane half h, elem e -> K=8h+(e&7)+16*(e>>3)
  float v = (k < K) ? W[(size_t)n * stride + colOff + k] : 0.f;
  dst[idx] = (bf16_t)v;
}

// memory (B, TENC, ESPK): per-b B-matrix with K=s, N=e
__global__ void k_memswz(const float* __restrict__ mem, bf16_t* __restrict__ dst) {
  int idx = blockIdx.x * blockDim.x + threadIdx.x;
  const int perB = 56 * 8 * 512;
  if (idx >= B_ * perB) return;
  int b = idx / perB, r = idx % perB;
  int e = r & 15, lane = (r >> 4) & 31, tile = r >> 9;
  int kt = tile % 8, nt = tile / 8;
  int n = nt * 16 + (lane & 15);
  int k = kt * 32 + ((lane >> 4) << 3) + (e & 7) + ((e >> 3) << 4);
  dst[idx] = (bf16_t)mem[((size_t)b * TENC + k) * ESPK + n];
}

__global__ void k_bias(const float* bi1, const float* bh1, const float* bi2,
                       const float* bh2, float* bs1, float* bs2) {
  int i = blockIdx.x * blockDim.x + threadIdx.x;
  if (i < 4096) { bs1[i] = bi1[i] + bh1[i]; bs2[i] = bi2[i] + bh2[i]; }
}

__global__ void k_pe(float* pe) {
  int i = blockIdx.x * blockDim.x + threadIdx.x;
  if (i >= MAXLEN * POSD) return;
  int p = i / POSD, d = i % POSD;
  float div = expf(-(float)(d & ~1) * (logf(10000.f) / (float)POSD));
  float ang = (float)p * div;
  pe[i] = (d & 1) ? cosf(ang) : sinf(ang);
}

__global__ void k_cvec(const float* __restrict__ dur, float* __restrict__ cvec) {
  __shared__ float s[TENC];
  int b = blockIdx.x, i = threadIdx.x;
  float d = dur[b * TENC + i];
  s[i] = d; __syncthreads();
  for (int off = 1; off < TENC; off <<= 1) {     // Hillis-Steele inclusive scan
    float v = (i >= off) ? s[i - off] : 0.f;
    __syncthreads();
    s[i] += v;
    __syncthreads();
  }
  cvec[b * TENC + i] = s[i] - d * 0.5f;
}

// ---------------- prenet (parallel over all 800 steps) ----------------
__global__ void k_prenet(const float* __restrict__ dec, const float* __restrict__ Wp1,
                         const float* __restrict__ Wp2, const float* __restrict__ u1,
                         const float* __restrict__ u2, bf16_t* __restrict__ inp1) {
  int t = blockIdx.x / B_, b = blockIdx.x % B_, j = threadIdx.x;
  __shared__ float xin[NMEL];
  __shared__ float x1[PRE_];
  if (j < NMEL)
    xin[j] = (t == 0) ? 0.f : dec[((size_t)b * NMEL + j) * TDEC + (t - 1)];
  __syncthreads();
  float a = 0.f;
  for (int m = 0; m < NMEL; m++) a += xin[m] * Wp1[j * NMEL + m];
  a = fmaxf(a, 0.f) * ((u1[((size_t)t * B_ + b) * PRE_ + j] >= 0.5f) ? 2.f : 0.f);
  x1[j] = a; __syncthreads();
  float a2 = 0.f;
  for (int m = 0; m < PRE_; m++) a2 += x1[m] * Wp2[j * PRE_ + m];
  a2 = fmaxf(a2, 0.f) * ((u2[((size_t)t * B_ + b) * PRE_ + j] >= 0.5f) ? 2.f : 0.f);
  inp1[(size_t)t * (2 * KT_IN1 * 512) + a_idx(b >> 4, j >> 5, KT_IN1, b & 15, j & 31)]
      = (bf16_t)a2;
}

// ---------------- attention weights (parallel; state-independent) ----------------
__global__ void k_attn(const float* __restrict__ rng, const float* __restrict__ cvec,
                       float* __restrict__ outA, bf16_t* __restrict__ wsw) {
  __shared__ float red[TENC];
  int b = blockIdx.x / TDEC, t = blockIdx.x % TDEC, s = threadIdx.x;
  float sig = rng[b * TENC + s] + 1e-5f;
  float coef = rsqrtf(2.f * PI_F * sig * sig);
  float dlt = (float)t - cvec[b * TENC + s];
  float prob = coef * expf(-dlt * dlt / (2.f * sig * sig)) + 1e-5f;
  red[s] = prob; __syncthreads();
  for (int off = 128; off > 0; off >>= 1) {
    if (s < off) red[s] += red[s + off];
    __syncthreads();
  }
  float w = prob / red[0];
  outA[((size_t)b * TDEC + t) * TENC + s] = w;     // alignments output (B,T,S)
  wsw[(size_t)b * (50 * 8 * 512) + a_idx(t >> 4, s >> 5, 8, t & 15, s & 31)]
      = (bf16_t)w;
}

// ---------------- context GEMM: per-b  (800x256)@(256x896), WMMA bf16 ----------------
__global__ void k_ctx(const bf16_t* __restrict__ wsw, const bf16_t* __restrict__ msw,
                      bf16_t* __restrict__ inp1) {
  int lane = threadIdx.x & 31;
  int gw = (blockIdx.x * blockDim.x + threadIdx.x) >> 5;
  const int TILES = 50 * 56;
  if (gw >= B_ * TILES) return;
  int b = gw / TILES, r = gw % TILES;
  int mt = r / 56, nt = r % 56;
  const bf16_t* A  = wsw + (size_t)b * 50 * 8 * 512 + (size_t)mt * 8 * 512;
  const bf16_t* Bm = msw + (size_t)b * 56 * 8 * 512 + (size_t)nt * 8 * 512;
  v8f acc = {};
  for (int kt = 0; kt < 8; kt++)
    acc = wmma_bf16(ld_frag(A + kt * 512, lane), ld_frag(Bm + kt * 512, lane), acc);
  int n0 = lane & 15, mh = (lane >> 4) * 8;
  for (int e = 0; e < 8; e++) {
    int t = mt * 16 + e + mh;
    int f = 256 + nt * 16 + n0;                 // ctx occupies inp1 features 256..1151
    inp1[(size_t)t * (2 * KT_IN1 * 512) + a_idx(b >> 4, f >> 5, KT_IN1, b & 15, f & 31)]
        = (bf16_t)acc[e];
  }
}

// ---------------- positional encodings into inp1 (features 1152..1279) ----------------
__global__ void k_pos(const int* __restrict__ dfi, const float* __restrict__ pe,
                      bf16_t* __restrict__ inp1) {
  int idx = blockIdx.x * blockDim.x + threadIdx.x;
  if (idx >= TDEC * B_ * POSD) return;
  int d = idx % POSD, tb = idx / POSD;
  int b = tb % B_, t = tb / B_;
  int pos = dfi[b * TDEC + t];
  float v = pe[pos * POSD + d];
  int f = 1152 + d;
  inp1[(size_t)t * (2 * KT_IN1 * 512) + a_idx(b >> 4, f >> 5, KT_IN1, b & 15, f & 31)]
      = (bf16_t)v;
}

// ---------------- mel partial: ctx_pe @ W_proj[:,1024:]  (parallel, WMMA) ----------------
__global__ void k_melp(const bf16_t* __restrict__ inp1, const bf16_t* __restrict__ Wpc,
                       const float* __restrict__ bproj, float* __restrict__ melp) {
  int lane = threadIdx.x & 31;
  int gw = (blockIdx.x * blockDim.x + threadIdx.x) >> 5;
  const int TILES = 2 * 5;
  if (gw >= TDEC * TILES) return;
  int t = gw / TILES, r = gw % TILES;
  int mt = r / 5, nt = r % 5;
  // ctx_pe == inp1 features 256..1279 == k-tiles 8..39 (matches W_proj col order)
  const bf16_t* A = inp1 + (size_t)t * (2 * KT_IN1 * 512) + ((size_t)mt * KT_IN1 + 8) * 512;
  const bf16_t* Bw = Wpc + (size_t)nt * 32 * 512;
  v8f acc = {};
  for (int kt = 0; kt < 32; kt++)
    acc = wmma_bf16(ld_frag(A + kt * 512, lane), ld_frag(Bw + kt * 512, lane), acc);
  int n0 = lane & 15, mh = (lane >> 4) * 8;
  for (int e = 0; e < 8; e++) {
    int b = mt * 16 + e + mh, j = nt * 16 + n0;
    melp[((size_t)t * B_ + b) * NMEL + j] = acc[e] + bproj[j];
  }
}

// ---------------- global barrier (sense reversal) ----------------
__device__ __forceinline__ void gbar(unsigned* cnt, unsigned* gen, int n) {
  __syncthreads();
  if (threadIdx.x == 0) {
    __threadfence();
    unsigned g = __hip_atomic_load(gen, __ATOMIC_RELAXED, __HIP_MEMORY_SCOPE_AGENT);
    unsigned p = __hip_atomic_fetch_add(cnt, 1u, __ATOMIC_ACQ_REL, __HIP_MEMORY_SCOPE_AGENT);
    if (p == (unsigned)(n - 1)) {
      __hip_atomic_store(cnt, 0u, __ATOMIC_RELAXED, __HIP_MEMORY_SCOPE_AGENT);
      __hip_atomic_store(gen, g + 1u, __ATOMIC_RELEASE, __HIP_MEMORY_SCOPE_AGENT);
    } else {
      while (__hip_atomic_load(gen, __ATOMIC_ACQUIRE, __HIP_MEMORY_SCOPE_AGENT) == g)
        __builtin_amdgcn_s_sleep(2);
    }
    __threadfence();
  }
  __syncthreads();
}

__global__ void k_zero(unsigned* p) { p[0] = 0u; p[1] = 0u; }

// LSTM cell on C-fragment registers; writes h (bf16, A-swizzled) and c (fp32)
__device__ __forceinline__ void lstm_cell_store(v8f gi, v8f gf, v8f gg, v8f go,
                                                const float* __restrict__ bs,
                                                float* __restrict__ c,
                                                bf16_t* __restrict__ hcur,
                                                int mt, int colbase, int lane) {
  int n0 = lane & 15, mh = (lane >> 4) * 8;
  int col = colbase + n0;
  float bi = bs[col], bf = bs[1024 + col], bg = bs[2048 + col], bo = bs[3072 + col];
  for (int e = 0; e < 8; e++) {
    int b = mt * 16 + e + mh;
    float i_ = gi[e] + bi, f_ = gf[e] + bf, g_ = gg[e] + bg, o_ = go[e] + bo;
    size_t ci = (size_t)b * 1024 + col;
    float cn = sigm(f_) * c[ci] + sigm(i_) * tanhf(g_);
    c[ci] = cn;
    float hn = sigm(o_) * tanhf(cn);
    hcur[a_idx(mt, col >> 5, KT_H, b & 15, col & 31)] = (bf16_t)hn;
  }
}

// ---------------- persistent sequential kernel (16 WGs, 8 waves each) ----------------
__global__ __launch_bounds__(256, 1) void k_seq(
    const bf16_t* __restrict__ inp1, const bf16_t* __restrict__ Wih1,
    const bf16_t* __restrict__ Whh1, const bf16_t* __restrict__ Wih2,
    const bf16_t* __restrict__ Whh2, const bf16_t* __restrict__ Wph,
    const float* __restrict__ bs1, const float* __restrict__ bs2,
    const float* __restrict__ melp, bf16_t* __restrict__ h1b,
    bf16_t* __restrict__ h2b, float* __restrict__ c1, float* __restrict__ c2,
    unsigned* bar, float* __restrict__ outMel) {
  const int wg = blockIdx.x, tid = threadIdx.x;
  const int lane = tid & 31, w = tid >> 5;
  const int mt = w & 1, ntl = w >> 1;            // wave owns (m-tile, 16 h-cols)
  const int colbase = wg * 64 + ntl * 16;
  const int ntg = wg * 4 + ntl;                  // n-tile within 1024 (0..63)
  const int HB = 2 * KT_H * 512;                 // elems per parity buffer

  for (int i = wg * 256 + tid; i < 2 * HB; i += NWG * 256) {
    h1b[i] = (bf16_t)0.f; h2b[i] = (bf16_t)0.f;
  }
  for (int i = wg * 256 + tid; i < 32 * 1024; i += NWG * 256) { c1[i] = 0.f; c2[i] = 0.f; }
  gbar(bar, bar + 1, NWG);

  for (int t = 0; t < TDEC; t++) {
    int par = t & 1;
    const bf16_t* h1prev = h1b + (par ^ 1) * HB;
    const bf16_t* h2prev = h2b + (par ^ 1) * HB;
    bf16_t* h1cur = h1b + par * HB;
    bf16_t* h2cur = h2b + par * HB;

    // ---- LSTM1: g = inp1[t] @ W_ih1^T + h1 @ W_hh1^T  (4 gates per wave) ----
    v8f a0 = {}, a1 = {}, a2 = {}, a3 = {};
    const bf16_t* Ain = inp1 + (size_t)t * (2 * KT_IN1 * 512) + (size_t)mt * KT_IN1 * 512;
    for (int kt = 0; kt < KT_IN1; kt++) {
      v16bf af = ld_frag(Ain + kt * 512, lane);
      a0 = wmma_bf16(af, ld_frag(Wih1 + ((size_t)(ntg)        * KT_IN1 + kt) * 512, lane), a0);
      a1 = wmma_bf16(af, ld_frag(Wih1 + ((size_t)(64  + ntg)  * KT_IN1 + kt) * 512, lane), a1);
      a2 = wmma_bf16(af, ld_frag(Wih1 + ((size_t)(128 + ntg)  * KT_IN1 + kt) * 512, lane), a2);
      a3 = wmma_bf16(af, ld_frag(Wih1 + ((size_t)(192 + ntg)  * KT_IN1 + kt) * 512, lane), a3);
    }
    const bf16_t* Ah1 = h1prev + (size_t)mt * KT_H * 512;
    for (int kt = 0; kt < KT_H; kt++) {
      v16bf af = ld_frag(Ah1 + kt * 512, lane);
      a0 = wmma_bf16(af, ld_frag(Whh1 + ((size_t)(ntg)        * KT_H + kt) * 512, lane), a0);
      a1 = wmma_bf16(af, ld_frag(Whh1 + ((size_t)(64  + ntg)  * KT_H + kt) * 512, lane), a1);
      a2 = wmma_bf16(af, ld_frag(Whh1 + ((size_t)(128 + ntg)  * KT_H + kt) * 512, lane), a2);
      a3 = wmma_bf16(af, ld_frag(Whh1 + ((size_t)(192 + ntg)  * KT_H + kt) * 512, lane), a3);
    }
    lstm_cell_store(a0, a1, a2, a3, bs1, c1, h1cur, mt, colbase, lane);
    gbar(bar, bar + 1, NWG);

    // ---- LSTM2: g = h1 @ W_ih2^T + h2 @ W_hh2^T ----
    a0 = (v8f){}; a1 = (v8f){}; a2 = (v8f){}; a3 = (v8f){};
    const bf16_t* A21 = h1cur + (size_t)mt * KT_H * 512;
    for (int kt = 0; kt < KT_H; kt++) {
      v16bf af = ld_frag(A21 + kt * 512, lane);
      a0 = wmma_bf16(af, ld_frag(Wih2 + ((size_t)(ntg)        * KT_H + kt) * 512, lane), a0);
      a1 = wmma_bf16(af, ld_frag(Wih2 + ((size_t)(64  + ntg)  * KT_H + kt) * 512, lane), a1);
      a2 = wmma_bf16(af, ld_frag(Wih2 + ((size_t)(128 + ntg)  * KT_H + kt) * 512, lane), a2);
      a3 = wmma_bf16(af, ld_frag(Wih2 + ((size_t)(192 + ntg)  * KT_H + kt) * 512, lane), a3);
    }
    const bf16_t* A22 = h2prev + (size_t)mt * KT_H * 512;
    for (int kt = 0; kt < KT_H; kt++) {
      v16bf af = ld_frag(A22 + kt * 512, lane);
      a0 = wmma_bf16(af, ld_frag(Whh2 + ((size_t)(ntg)        * KT_H + kt) * 512, lane), a0);
      a1 = wmma_bf16(af, ld_frag(Whh2 + ((size_t)(64  + ntg)  * KT_H + kt) * 512, lane), a1);
      a2 = wmma_bf16(af, ld_frag(Whh2 + ((size_t)(128 + ntg)  * KT_H + kt) * 512, lane), a2);
      a3 = wmma_bf16(af, ld_frag(Whh2 + ((size_t)(192 + ntg)  * KT_H + kt) * 512, lane), a3);
    }
    lstm_cell_store(a0, a1, a2, a3, bs2, c2, h2cur, mt, colbase, lane);
    gbar(bar, bar + 1, NWG);

    // ---- mel = h2 @ W_proj[:,:1024]^T + melp[t]  (10 tiles on first 10 waves) ----
    int gwv = wg * 8 + w;
    if (gwv < 10) {
      int mmt = gwv & 1, nt = gwv >> 1;
      v8f acc = {};
      const bf16_t* Am = h2cur + (size_t)mmt * KT_H * 512;
      for (int kt = 0; kt < KT_H; kt++)
        acc = wmma_bf16(ld_frag(Am + kt * 512, lane),
                        ld_frag(Wph + ((size_t)nt * KT_H + kt) * 512, lane), acc);
      int n0 = lane & 15, mh = (lane >> 4) * 8;
      for (int e = 0; e < 8; e++) {
        int b = mmt * 16 + e + mh, j = nt * 16 + n0;
        outMel[((size_t)b * NMEL + j) * TDEC + t] =
            acc[e] + melp[((size_t)t * B_ + b) * NMEL + j];
      }
    }
    // mel reads h2cur[par]; next write to this buffer is step t+2 phase2,
    // which is >=2 barriers away, so no extra barrier needed.
  }
}

// ---------------- host launcher ----------------
extern "C" void kernel_launch(void* const* d_in, const int* in_sizes, int n_in,
                              void* d_out, int out_size, void* d_ws, size_t ws_size,
                              hipStream_t stream) {
  const float* memory = (const float*)d_in[0];
  const float* dur    = (const float*)d_in[1];
  const int*   dfi    = (const int*)d_in[2];
  const float* rng    = (const float*)d_in[3];
  const float* decin  = (const float*)d_in[4];
  // d_in[5] memory_lengths: unused by the reference
  const float* Wp1  = (const float*)d_in[6];
  const float* Wp2  = (const float*)d_in[7];
  const float* Wih1 = (const float*)d_in[8];
  const float* Whh1 = (const float*)d_in[9];
  const float* bih1 = (const float*)d_in[10];
  const float* bhh1 = (const float*)d_in[11];
  const float* Wih2 = (const float*)d_in[12];
  const float* Whh2 = (const float*)d_in[13];
  const float* bih2 = (const float*)d_in[14];
  const float* bhh2 = (const float*)d_in[15];
  const float* Wproj = (const float*)d_in[16];
  const float* bproj = (const float*)d_in[17];
  const float* u1 = (const float*)d_in[18];
  const float* u2 = (const float*)d_in[19];

  float* outMel   = (float*)d_out;
  float* outAlign = outMel + (size_t)B_ * NMEL * TDEC;

  char* ws = (char*)d_ws;
  size_t off = 0;
  auto alloc = [&](size_t bytes) -> char* {
    char* p = ws + off;
    off += (bytes + 255) & ~(size_t)255;
    return p;
  };
  bf16_t* wih1s = (bf16_t*)alloc(256ull * 40 * 512 * 2);
  bf16_t* whh1s = (bf16_t*)alloc(256ull * 32 * 512 * 2);
  bf16_t* wih2s = (bf16_t*)alloc(256ull * 32 * 512 * 2);
  bf16_t* whh2s = (bf16_t*)alloc(256ull * 32 * 512 * 2);
  bf16_t* wphs  = (bf16_t*)alloc(5ull * 32 * 512 * 2);
  bf16_t* wpcs  = (bf16_t*)alloc(5ull * 32 * 512 * 2);
  bf16_t* msw   = (bf16_t*)alloc(32ull * 56 * 8 * 512 * 2);
  bf16_t* inp1  = (bf16_t*)alloc(800ull * 2 * 40 * 512 * 2);
  bf16_t* wsw   = (bf16_t*)alloc(32ull * 50 * 8 * 512 * 2);
  float*  melp  = (float*) alloc(800ull * 32 * 80 * 4);
  bf16_t* h1b   = (bf16_t*)alloc(2ull * 2 * 32 * 512 * 2);
  bf16_t* h2b   = (bf16_t*)alloc(2ull * 2 * 32 * 512 * 2);
  float*  c1    = (float*) alloc(32ull * 1024 * 4);
  float*  c2    = (float*) alloc(32ull * 1024 * 4);
  float*  bs1   = (float*) alloc(4096ull * 4);
  float*  bs2   = (float*) alloc(4096ull * 4);
  float*  petab = (float*) alloc((size_t)MAXLEN * POSD * 4);
  float*  cvec  = (float*) alloc(32ull * 256 * 4);
  unsigned* bar = (unsigned*)alloc(256);

  const int T = 256;
  // weight swizzles (bf16 B-fragment layout)
  k_bswz<<<(256 * 40 * 512 + T - 1) / T, T, 0, stream>>>(Wih1, wih1s, 4096, 1280, 1280, 0, 40);
  k_bswz<<<(256 * 32 * 512 + T - 1) / T, T, 0, stream>>>(Whh1, whh1s, 4096, 1024, 1024, 0, 32);
  k_bswz<<<(256 * 32 * 512 + T - 1) / T, T, 0, stream>>>(Wih2, wih2s, 4096, 1024, 1024, 0, 32);
  k_bswz<<<(256 * 32 * 512 + T - 1) / T, T, 0, stream>>>(Whh2, whh2s, 4096, 1024, 1024, 0, 32);
  k_bswz<<<(5 * 32 * 512 + T - 1) / T, T, 0, stream>>>(Wproj, wphs, 80, 1024, 2048, 0, 32);
  k_bswz<<<(5 * 32 * 512 + T - 1) / T, T, 0, stream>>>(Wproj, wpcs, 80, 1024, 2048, 1024, 32);
  k_memswz<<<(32 * 56 * 8 * 512 + T - 1) / T, T, 0, stream>>>(memory, msw);
  k_bias<<<16, T, 0, stream>>>(bih1, bhh1, bih2, bhh2, bs1, bs2);
  k_pe<<<(MAXLEN * POSD + T - 1) / T, T, 0, stream>>>(petab);
  k_cvec<<<32, TENC, 0, stream>>>(dur, cvec);

  // parallel precompute
  k_prenet<<<TDEC * B_, PRE_, 0, stream>>>(decin, Wp1, Wp2, u1, u2, inp1);
  k_attn<<<B_ * TDEC, TENC, 0, stream>>>(rng, cvec, outAlign, wsw);
  k_ctx<<<(B_ * 50 * 56 + 7) / 8, T, 0, stream>>>(wsw, msw, inp1);
  k_pos<<<(TDEC * B_ * POSD + T - 1) / T, T, 0, stream>>>(dfi, petab, inp1);
  k_melp<<<(TDEC * 10 + 7) / 8, T, 0, stream>>>(inp1, wpcs, bproj, melp);

  // sequential recurrence (persistent, grid-wide barriers)
  k_zero<<<1, 1, 0, stream>>>(bar);
  k_seq<<<NWG, T, 0, stream>>>(inp1, wih1s, whh1s, wih2s, whh2s, wphs, bs1, bs2,
                               melp, h1b, h2b, c1, c2, bar, outMel);
}
